// TransLayer_11063835754986
// MI455X (gfx1250) — matrix operand order repaired and checked by reading, not compiled, not run on hardware
//
#include <hip/hip_runtime.h>

#define DEV __device__ __forceinline__

typedef __attribute__((ext_vector_type(16))) _Float16 v16h;
typedef __attribute__((ext_vector_type(8)))  float    v8f;
typedef __attribute__((ext_vector_type(2)))  float    v2f;
typedef __attribute__((ext_vector_type(4)))  unsigned int v4u;
typedef __attribute__((ext_vector_type(8)))  int      v8i;
typedef __attribute__((ext_vector_type(4)))  int      v4i;

#define B_    4
#define N_    8192
#define D_    512
#define H_    8
#define DH_   64
#define M_    256
#define L_    32
#define BH_   32
#define ROWS_ (B_*N_)      /* 32768 */
#define QKVC_ 1536
#define KER_  33

// ---------------- wave helpers ----------------
DEV float wred_sum(float v){            // reduce across all 32 lanes
  #pragma unroll
  for(int m=16;m>=1;m>>=1) v += __shfl_xor(v,m,32);
  return v;
}
DEV float hred_sum(float v){            // reduce across 16-lane half (C-tile row)
  #pragma unroll
  for(int m=8;m>=1;m>>=1) v += __shfl_xor(v,m,32);
  return v;
}
DEV float hred_max(float v){
  #pragma unroll
  for(int m=8;m>=1;m>>=1) v = fmaxf(v,__shfl_xor(v,m,32));
  return v;
}

// ---------------- WMMA wrappers + tile loaders (contiguous k order: element s of lane l holds k = 16*(l/16) + s) ----------------
DEV v8f wmma16(v16h a, v16h b, v8f c){
  return __builtin_amdgcn_wmma_f32_16x16x32_f16(false, a, false, b, (short)0, c, false, false);
}

// A tile 16x32 f16 from row-major memory (row stride ld elements)
DEV v16h load_a16(const _Float16* p, int ld){
  int lane = threadIdx.x & 31;
  int m  = lane & 15;
  int kb = (lane < 16) ? 0 : 16;
  const _Float16* r = p + (size_t)m*ld;
  v16h a;
  #pragma unroll
  for(int v=0; v<4; ++v){ a[2*v] = r[kb+2*v]; a[2*v+1] = r[kb+2*v+1]; }
  #pragma unroll
  for(int v=4; v<8; ++v){ int k = kb + 2*v; a[2*v] = r[k]; a[2*v+1] = r[k+1]; }
  return a;
}

// B tile 32x16 f16, memory is row-major KxN (element (k,n) at p[k*ld+n])
DEV v16h load_b16(const _Float16* p, int ld){
  int lane = threadIdx.x & 31;
  int n  = lane & 15;
  int kb = (lane < 16) ? 0 : 16;
  v16h b;
  #pragma unroll
  for(int v=0; v<8; ++v){
    b[2*v]   = p[(size_t)(kb+2*v)*ld + n];
    b[2*v+1] = p[(size_t)(kb+2*v+1)*ld + n];
  }
  return b;
}

// B tile 32x16 f16 where memory holds the transpose: element (k,n) at p[n*ld+k]
DEV v16h load_bt16(const _Float16* p, int ld){
  int lane = threadIdx.x & 31;
  int n  = lane & 15;
  int kb = (lane < 16) ? 0 : 16;
  const _Float16* r = p + (size_t)n*ld + kb;
  v16h b;
  #pragma unroll
  for(int v=0; v<8; ++v){ b[2*v] = r[2*v]; b[2*v+1] = r[2*v+1]; }
  return b;
}

// store f32 16x16 C tile to row-major memory
DEV void store_c16(float* p, int ld, v8f c){
  int lane = threadIdx.x & 31;
  int n  = lane & 15;
  int mb = (lane < 16) ? 0 : 8;
  #pragma unroll
  for(int r=0;r<8;++r) p[(size_t)(mb+r)*ld + n] = c[r];
}

// in-place row softmax over a 16-row x 256-col block held as 16 C tiles
DEV void softmax16(v8f (&acc)[16]){
  #pragma unroll
  for(int r=0;r<8;++r){
    float mx = acc[0][r];
    #pragma unroll
    for(int t=1;t<16;++t) mx = fmaxf(mx, acc[t][r]);
    mx = hred_max(mx);
    float s = 0.f;
    #pragma unroll
    for(int t=0;t<16;++t){ acc[t][r] = __expf(acc[t][r]-mx); s += acc[t][r]; }
    s = hred_sum(s);
    float inv = 1.f/s;
    #pragma unroll
    for(int t=0;t<16;++t) acc[t][r] *= inv;
  }
}

// ---------------- TDM: DMA a (rows x tileW) f16 panel global -> LDS ----------------
// D# packing per CDNA5 ISA (2D tensor, groups 2/3 zero).
DEV void tdm_load_panel_f16(_Float16* ldsDst, const _Float16* gsrc,
                            unsigned tensorW, unsigned rows, unsigned tileW){
#if __has_builtin(__builtin_amdgcn_tensor_load_to_lds)
  unsigned long long ga = (unsigned long long)(size_t)gsrc;   // byte address
  unsigned lds_off = (unsigned)(size_t)ldsDst;                // LDS byte offset (low 32 bits)
  v4u g0;
  g0[0] = 1u;                                           // count=1, load, no gather
  g0[1] = lds_off;                                      // lds_addr
  g0[2] = (unsigned)(ga & 0xffffffffu);                 // global_addr[31:0]
  g0[3] = (unsigned)((ga >> 32) & 0x01ffffffu)          // global_addr[56:32]
          | 0x80000000u;                                // type=2 ("image")
  v8i g1;
  g1[0] = (int)(1u << 16);                              // data_size=1 (2 bytes)
  g1[1] = (int)((tensorW & 0xffffu) << 16);             // tensor_dim0[15:0]
  g1[2] = (int)((tensorW >> 16) | ((rows & 0xffffu) << 16));   // td0 hi | tensor_dim1 lo
  g1[3] = (int)((rows >> 16) | (tileW << 16));          // td1 hi | tile_dim0
  g1[4] = (int)(rows & 0xffffu);                        // tile_dim1, tile_dim2=0
  g1[5] = (int)tensorW;                                 // tensor_dim0_stride[31:0]
  g1[6] = 0;                                            // stride0 hi, stride1 lo
  g1[7] = 0;                                            // stride1 hi
  v4i z = {0,0,0,0};
#if __clang_major__ >= 23
  v8i z8 = {0,0,0,0,0,0,0,0};
  __builtin_amdgcn_tensor_load_to_lds(g0, g1, z, z, z8, 0);
#else
  __builtin_amdgcn_tensor_load_to_lds(g0, g1, z, z, 0);
#endif
  __builtin_amdgcn_s_wait_tensorcnt((short)0);
#else
  // fallback cooperative copy by one wave
  int lane = threadIdx.x & 31;
  for(unsigned r = (unsigned)lane; r < rows; r += 32)
    for(unsigned c2 = 0; c2 < tileW; ++c2)
      ldsDst[(size_t)r*tileW + c2] = gsrc[(size_t)r*tensorW + c2];
#endif
}

// ---------------- kernels ----------------

// LayerNorm: one wave per token row, output f16
__global__ void k_ln(const float* __restrict__ x, const float* __restrict__ g,
                     const float* __restrict__ bt, _Float16* __restrict__ xn){
  int row  = blockIdx.x*8 + (threadIdx.x>>5);
  int lane = threadIdx.x & 31;
  const float* xr = x + (size_t)row*D_;
  float s = 0.f;
  #pragma unroll
  for(int i=0;i<16;++i) s += xr[lane + i*32];
  float mu = wred_sum(s) * (1.f/D_);
  float v = 0.f;
  #pragma unroll
  for(int i=0;i<16;++i){ float d = xr[lane+i*32]-mu; v += d*d; }
  float rs = rsqrtf(wred_sum(v)*(1.f/D_) + 1e-5f);
  _Float16* o = xn + (size_t)row*D_;
  #pragma unroll
  for(int i=0;i<16;++i){
    int c = lane + i*32;
    o[c] = (_Float16)((xr[c]-mu)*rs*g[c] + bt[c]);
  }
}

__global__ void k_cvt(const float* __restrict__ in, _Float16* __restrict__ o, int n){
  int idx = blockIdx.x*256 + threadIdx.x;
  if(idx < n) o[idx] = (_Float16)in[idx];
}

// qkv = xn @ w_qkv; B panel (512x64 f16 = 64KB) staged in LDS via TDM, shared by 8 waves.
// Block: nb = blockIdx.x % 24 (64-col panel), mgrp = blockIdx.x / 24 (8 row-blocks).
__global__ void __launch_bounds__(256) k_qkv(const _Float16* __restrict__ xn,
                                             const _Float16* __restrict__ wq,
                                             _Float16* __restrict__ q,
                                             _Float16* __restrict__ k,
                                             _Float16* __restrict__ v){
  __shared__ _Float16 Bp[D_*64];          // 512 x 64 f16 panel = 64 KB
  int nb   = blockIdx.x % 24;
  int mgrp = blockIdx.x / 24;
  int wid  = threadIdx.x >> 5;
  if(wid == 0) tdm_load_panel_f16(Bp, wq + nb*64, QKVC_, D_, 64);
  __syncthreads();

  int mblk = mgrp*8 + wid;
  int row0 = mblk*16;
  int b    = row0 / N_;
  int nn0  = row0 % N_;
  v8f acc[4] = {};
  for(int kc=0;kc<D_;kc+=32){
    v16h a = load_a16(xn + (size_t)row0*D_ + kc, D_);
    #pragma unroll
    for(int t=0;t<4;++t)
      acc[t] = wmma16(a, load_b16(Bp + (size_t)kc*64 + t*16, 64), acc[t]);
  }
  int lane = threadIdx.x&31, n = lane&15, mb = (lane<16)?0:8;
  #pragma unroll
  for(int t=0;t<4;++t){
    int cb  = nb*64 + t*16;
    int sec = cb>>9, hh = (cb>>6)&7, db = cb&63;
    _Float16* dst = (sec==0)? q : ((sec==1)? k : v);
    float sc = (sec==0)? 0.125f : 1.f;
    #pragma unroll
    for(int r=0;r<8;++r){
      int nn = nn0 + mb + r;
      dst[(((size_t)(b*H_+hh))*N_ + nn)*DH_ + db + n] = (_Float16)(acc[t][r]*sc);
    }
  }
}

// landmark means: one wave per output row (two tensors: q_l, k_l)
__global__ void k_land(const _Float16* __restrict__ q, const _Float16* __restrict__ k,
                       _Float16* __restrict__ ql, _Float16* __restrict__ kl){
  int gid = blockIdx.x*8 + (threadIdx.x>>5);
  int tensor = gid>>13;
  int rid = gid & 8191;
  int bh = rid>>8, m = rid&255;
  const _Float16* src = tensor? k : q;
  _Float16* dst = tensor? kl : ql;
  int lane = threadIdx.x&31;
  int d0 = lane*2;
  float s0=0.f, s1=0.f;
  const _Float16* base = src + ((size_t)bh*N_ + m*L_)*DH_ + d0;
  #pragma unroll
  for(int r=0;r<L_;++r){ s0 += (float)base[(size_t)r*DH_]; s1 += (float)base[(size_t)r*DH_+1]; }
  _Float16* o = dst + ((size_t)bh*M_ + m)*DH_ + d0;
  o[0] = (_Float16)(s0*(1.f/L_));
  o[1] = (_Float16)(s1*(1.f/L_));
}

// a2 = softmax(q_l @ k_l^T), f32 output (bh,256,256). One wave per 16-row block.
__global__ void k_a2(const _Float16* __restrict__ ql, const _Float16* __restrict__ kl,
                     float* __restrict__ a2){
  int gid  = blockIdx.x*4 + (threadIdx.x>>5);
  int bh   = gid>>4, mblk = gid&15;
  const _Float16* qb = ql + ((size_t)bh*M_ + mblk*16)*DH_;
  v16h a0 = load_a16(qb, DH_), a1 = load_a16(qb+32, DH_);
  v8f acc[16] = {};
  #pragma unroll
  for(int t=0;t<16;++t){
    const _Float16* kb = kl + ((size_t)bh*M_ + t*16)*DH_;
    acc[t] = wmma16(a0, load_bt16(kb,    DH_), acc[t]);
    acc[t] = wmma16(a1, load_bt16(kb+32, DH_), acc[t]);
  }
  softmax16(acc);
  float* C = a2 + (size_t)bh*M_*M_ + (size_t)(mblk*16)*M_;
  #pragma unroll
  for(int t=0;t<16;++t) store_c16(C + t*16, M_, acc[t]);
}

__global__ void k_init(float* s){ if(threadIdx.x<2) s[threadIdx.x]=0.f; }

// global max row-sum / max col-sum of a2 (all entries positive)
__global__ void k_rowcol(const float* __restrict__ a2, float* __restrict__ scal){
  int gid = blockIdx.x*8 + (threadIdx.x>>5);
  int bh = gid>>8, i = gid&255;
  int lane = threadIdx.x&31;
  const float* base = a2 + (size_t)bh*M_*M_;
  float rs=0.f, cs=0.f;
  #pragma unroll
  for(int jj=0;jj<8;++jj){
    int j = lane + jj*32;
    rs += base[(size_t)i*M_ + j];
    cs += base[(size_t)j*M_ + i];
  }
  rs = wred_sum(rs); cs = wred_sum(cs);
  if(lane==0){
    atomicMax((int*)scal,     __float_as_int(rs));
    atomicMax(((int*)scal)+1, __float_as_int(cs));
  }
}

// z0 = a2^T / (rmax*cmax)
__global__ void k_z0(const float* __restrict__ a2, const float* __restrict__ scal,
                     float* __restrict__ z){
  int idx = blockIdx.x*256 + threadIdx.x;
  int bh = idx >> 16;
  int r  = idx & 65535;
  int i  = r >> 8, j = r & 255;
  float inv = 1.f/(scal[0]*scal[1]);
  z[idx] = a2[(size_t)bh*65536 + (size_t)j*256 + i]*inv;
}

// out = alpha*in + cdiag*I (elementwise, 256x256 batches)
__global__ void k_axpyI(const float* __restrict__ in, float* __restrict__ o,
                        float alpha, float cdiag){
  int idx = blockIdx.x*256 + threadIdx.x;
  int i = (idx>>8)&255, j = idx&255;
  o[idx] = alpha*in[idx] + ((i==j)?cdiag:0.f);
}

// batched f32 GEMM via v_wmma_f32_16x16x4_f32: C = cdiag*I + alpha*(A@B)
__global__ void k_gemm_f32(const float* __restrict__ A, const float* __restrict__ B,
                           float* __restrict__ C, int tm, int tn, int K,
                           int lda, int ldb, int ldc,
                           long long sA, long long sB, long long sC,
                           float alpha, float cdiag, int total){
  int gid = blockIdx.x*(blockDim.x>>5) + (threadIdx.x>>5);
  if(gid >= total) return;
  int tpb = tm*tn;
  int bh = gid / tpb, t = gid % tpb;
  int mt = t / tn, nt = t % tn;
  int lane = threadIdx.x&31;
  int mn  = lane & 15;
  int kb  = (lane<16)? 0 : 2;
  const float* Ar = A + (size_t)bh*sA + (size_t)(mt*16 + mn)*lda;
  const float* Bb = B + (size_t)bh*sB + nt*16 + mn;
  v8f acc = {};
  for(int k0=0;k0<K;k0+=4){
    v2f av, bv;
    av[0] = Ar[k0+kb];                 av[1] = Ar[k0+kb+1];
    bv[0] = Bb[(size_t)(k0+kb)*ldb];   bv[1] = Bb[(size_t)(k0+kb+1)*ldb];
    acc = __builtin_amdgcn_wmma_f32_16x16x4_f32(false, av, false, bv, (short)0, acc, false, false);
  }
  int mb = (lane<16)? 0 : 8;
  float* Cb = C + (size_t)bh*sC + (size_t)(mt*16)*ldc + nt*16;
  #pragma unroll
  for(int r=0;r<8;++r){
    int gi = mt*16+mb+r, gj = nt*16+mn;
    Cb[(size_t)(mb+r)*ldc + mn] = alpha*acc[r] + ((gi==gj)? cdiag : 0.f);
  }
}

// a3v = softmax(q_l @ k^T) @ v  — flash-attention style online softmax, per-wave 16 q rows
__global__ void __launch_bounds__(128) k_a3v(const _Float16* __restrict__ ql,
                                             const _Float16* __restrict__ k,
                                             const _Float16* __restrict__ v,
                                             float* __restrict__ a3v){
  __shared__ _Float16 lds[4*16*32];
  int wid = threadIdx.x>>5;
  int gid = blockIdx.x*4 + wid;
  int bh  = gid>>4, mblk = gid&15;
  const _Float16* qb = ql + ((size_t)bh*M_ + mblk*16)*DH_;
  v16h a0 = load_a16(qb, DH_), a1 = load_a16(qb+32, DH_);
  v8f o[4] = {};
  float mrow[8], lrow[8];
  #pragma unroll
  for(int r=0;r<8;++r){ mrow[r] = -1e30f; lrow[r] = 0.f; }
  _Float16* Lw = lds + wid*(16*32);
  int lane = threadIdx.x&31, n = lane&15, mb = (lane<16)?0:8;

  for(int j0=0; j0<N_; j0+=32){
    const _Float16* kb = k + ((size_t)bh*N_ + j0)*DH_;
    v8f s0 = {}, s1 = {};
    s0 = wmma16(a0, load_bt16(kb,            DH_), s0);
    s0 = wmma16(a1, load_bt16(kb+32,         DH_), s0);
    s1 = wmma16(a0, load_bt16(kb+16*DH_,     DH_), s1);
    s1 = wmma16(a1, load_bt16(kb+16*DH_+32,  DH_), s1);
    float nm[8], sc[8];
    #pragma unroll
    for(int r=0;r<8;++r){
      float mx = hred_max(fmaxf(s0[r], s1[r]));
      nm[r] = fmaxf(mrow[r], mx);
      sc[r] = __expf(mrow[r]-nm[r]);
      mrow[r] = nm[r];
    }
    #pragma unroll
    for(int r=0;r<8;++r){
      s0[r] = __expf(s0[r]-nm[r]);
      s1[r] = __expf(s1[r]-nm[r]);
      lrow[r] = lrow[r]*sc[r] + hred_sum(s0[r]+s1[r]);
    }
    #pragma unroll
    for(int t=0;t<4;++t)
      #pragma unroll
      for(int r=0;r<8;++r) o[t][r] *= sc[r];
    // C-layout -> A-layout transpose through per-wave LDS tile
    #pragma unroll
    for(int r=0;r<8;++r){
      Lw[(mb+r)*32 + n]      = (_Float16)s0[r];
      Lw[(mb+r)*32 + 16 + n] = (_Float16)s1[r];
    }
    v16h ap = load_a16(Lw, 32);
    const _Float16* vb = v + ((size_t)bh*N_ + j0)*DH_;
    #pragma unroll
    for(int t=0;t<4;++t) o[t] = wmma16(ap, load_b16(vb + t*16, DH_), o[t]);
  }
  #pragma unroll
  for(int t=0;t<4;++t)
    #pragma unroll
    for(int r=0;r<8;++r) o[t][r] /= lrow[r];
  float* C = a3v + ((size_t)bh*M_ + mblk*16)*DH_;
  #pragma unroll
  for(int t=0;t<4;++t) store_c16(C + t*16, DH_, o[t]);
}

// out1 = softmax(q @ k_l^T) @ w2   (a1 fused, never materialized)
__global__ void __launch_bounds__(128) k_a1w(const _Float16* __restrict__ q,
                                             const _Float16* __restrict__ kl,
                                             const _Float16* __restrict__ w2h,
                                             float* __restrict__ out1){
  __shared__ _Float16 lds[4*16*32];
  int wid = threadIdx.x>>5;
  int gid = blockIdx.x*4 + wid;
  int bh  = gid>>9, mblk = gid&511;
  const _Float16* qb = q + ((size_t)bh*N_ + mblk*16)*DH_;
  v16h a0 = load_a16(qb, DH_), a1 = load_a16(qb+32, DH_);
  v8f acc[16] = {};
  #pragma unroll
  for(int t=0;t<16;++t){
    const _Float16* kb = kl + ((size_t)bh*M_ + t*16)*DH_;
    acc[t] = wmma16(a0, load_bt16(kb,    DH_), acc[t]);
    acc[t] = wmma16(a1, load_bt16(kb+32, DH_), acc[t]);
  }
  softmax16(acc);
  v8f o[4] = {};
  _Float16* Lw = lds + wid*(16*32);
  int lane = threadIdx.x&31, n = lane&15, mb = (lane<16)?0:8;
  #pragma unroll
  for(int c=0;c<8;++c){
    #pragma unroll
    for(int tt=0;tt<2;++tt)
      #pragma unroll
      for(int r=0;r<8;++r)
        Lw[(mb+r)*32 + tt*16 + n] = (_Float16)acc[2*c+tt][r];
    v16h ap = load_a16(Lw, 32);
    #pragma unroll
    for(int t=0;t<4;++t)
      o[t] = wmma16(ap, load_b16(w2h + (size_t)(c*32)*DH_ + t*16, DH_), o[t]);
  }
  float* C = out1 + ((size_t)bh*N_ + mblk*16)*DH_;
  #pragma unroll
  for(int t=0;t<4;++t) store_c16(C + t*16, DH_, o[t]);
}

// depthwise conv residual + reorder to (b, n, h*dh) f16
__global__ void k_conv(const float* __restrict__ out1, const _Float16* __restrict__ v,
                       const float* __restrict__ cw, _Float16* __restrict__ att){
  int idx = blockIdx.x*256 + threadIdx.x;
  int c   = idx & (D_-1);
  int row = idx >> 9;
  int b = row / N_, nn = row % N_;
  int hh = c>>6, d = c&63;
  int bh = b*H_ + hh;
  const _Float16* vb = v + (size_t)bh*N_*DH_ + d;
  float acc = 0.f;
  #pragma unroll
  for(int t=0;t<KER_;++t){
    int j = nn + t - (KER_/2);
    if(j>=0 && j<N_) acc += cw[hh*KER_+t]*(float)vb[(size_t)j*DH_];
  }
  acc += out1[((size_t)bh*N_ + nn)*DH_ + d];
  att[(size_t)row*D_ + c] = (_Float16)acc;
}

// final: d_out = x + att @ w_out + b_out; w_out panel staged in LDS via TDM
__global__ void __launch_bounds__(256) k_final(const _Float16* __restrict__ att,
                                               const _Float16* __restrict__ wo,
                                               const float* __restrict__ bo,
                                               const float* __restrict__ x,
                                               float* __restrict__ out){
  __shared__ _Float16 Bp[D_*64];          // 512 x 64 f16 panel = 64 KB
  int nb   = blockIdx.x & 7;
  int mgrp = blockIdx.x >> 3;
  int wid  = threadIdx.x >> 5;
  if(wid == 0) tdm_load_panel_f16(Bp, wo + nb*64, D_, D_, 64);
  __syncthreads();

  int mblk = mgrp*8 + wid;
  int row0 = mblk*16;
  v8f acc[4] = {};
  for(int kc=0;kc<D_;kc+=32){
    v16h a = load_a16(att + (size_t)row0*D_ + kc, D_);
    #pragma unroll
    for(int t=0;t<4;++t)
      acc[t] = wmma16(a, load_b16(Bp + (size_t)kc*64 + t*16, 64), acc[t]);
  }
  int lane = threadIdx.x&31, n = lane&15, mb = (lane<16)?0:8;
  #pragma unroll
  for(int t=0;t<4;++t){
    int cb = nb*64 + t*16 + n;
    #pragma unroll
    for(int r=0;r<8;++r){
      size_t ix = (size_t)(row0+mb+r)*D_ + cb;
      out[ix] = x[ix] + acc[t][r] + bo[cb];
    }
  }
}

// ---------------- host ----------------
static void launch_gemm(const float* A, const float* B, float* C,
                        int Mi, int Ni, int K, float alpha, float cdiag,
                        hipStream_t s){
  int tm = Mi/16, tn = Ni/16;
  int total = BH_*tm*tn;
  int blocks = (total + 7)/8;
  k_gemm_f32<<<blocks,256,0,s>>>(A,B,C,tm,tn,K, K, Ni, Ni,
                                 (long long)Mi*K, (long long)K*Ni, (long long)Mi*Ni,
                                 alpha, cdiag, total);
}

extern "C" void kernel_launch(void* const* d_in, const int* in_sizes, int n_in,
                              void* d_out, int out_size, void* d_ws, size_t ws_size,
                              hipStream_t stream) {
  (void)in_sizes; (void)n_in; (void)out_size; (void)ws_size;
  const float* x      = (const float*)d_in[0];
  const float* ln_w   = (const float*)d_in[1];
  const float* ln_b   = (const float*)d_in[2];
  const float* w_qkv  = (const float*)d_in[3];
  const float* w_out  = (const float*)d_in[4];
  const float* b_out  = (const float*)d_in[5];
  const float* conv_w = (const float*)d_in[6];
  float* outp = (float*)d_out;

  char* w = (char*)d_ws; size_t off = 0;
  auto alloc = [&](size_t bytes)->void*{
    void* p = w + off; off = (off + bytes + 255) & ~(size_t)255; return p;
  };
  _Float16* xn  = (_Float16*)alloc((size_t)ROWS_*D_*2);
  _Float16* wqh = (_Float16*)alloc((size_t)D_*QKVC_*2);
  _Float16* woh = (_Float16*)alloc((size_t)D_*D_*2);
  _Float16* qh  = (_Float16*)alloc((size_t)BH_*N_*DH_*2);
  _Float16* kh  = (_Float16*)alloc((size_t)BH_*N_*DH_*2);
  _Float16* vh  = (_Float16*)alloc((size_t)BH_*N_*DH_*2);
  _Float16* qlh = (_Float16*)alloc((size_t)BH_*M_*DH_*2);
  _Float16* klh = (_Float16*)alloc((size_t)BH_*M_*DH_*2);
  float* a2  = (float*)alloc((size_t)BH_*M_*M_*4);
  float* Z0  = (float*)alloc((size_t)BH_*M_*M_*4);
  float* Z1  = (float*)alloc((size_t)BH_*M_*M_*4);
  float* AZ  = (float*)alloc((size_t)BH_*M_*M_*4);
  float* U   = (float*)alloc((size_t)BH_*M_*M_*4);
  float* T   = (float*)alloc((size_t)BH_*M_*M_*4);
  float* a3v = (float*)alloc((size_t)BH_*M_*DH_*4);
  float* w2  = (float*)alloc((size_t)BH_*M_*DH_*4);
  _Float16* w2h = (_Float16*)alloc((size_t)BH_*M_*DH_*2);
  float* out1 = (float*)alloc((size_t)BH_*N_*DH_*4);
  _Float16* att = (_Float16*)alloc((size_t)ROWS_*D_*2);
  float* scal = (float*)alloc(256);

  k_ln  <<<ROWS_/8, 256, 0, stream>>>(x, ln_w, ln_b, xn);
  k_cvt <<<(D_*QKVC_)/256, 256, 0, stream>>>(w_qkv, wqh, D_*QKVC_);
  k_cvt <<<(D_*D_)/256,    256, 0, stream>>>(w_out, woh, D_*D_);
  k_qkv <<<6144, 256, 0, stream>>>(xn, wqh, qh, kh, vh);
  k_land<<<2048, 256, 0, stream>>>(qh, kh, qlh, klh);
  k_a2  <<<128, 128, 0, stream>>>(qlh, klh, a2);
  k_init<<<1, 32, 0, stream>>>(scal);
  k_rowcol<<<1024, 256, 0, stream>>>(a2, scal);
  k_z0  <<<(BH_*M_*M_)/256, 256, 0, stream>>>(a2, scal, Z0);

  float* zc = Z0; float* zn = Z1;
  for(int it=0; it<6; ++it){
    launch_gemm(a2, zc, AZ, 256, 256, 256, 1.0f,  0.0f, stream);      // AZ = a @ z
    k_axpyI<<<(BH_*M_*M_)/256, 256, 0, stream>>>(AZ, U, -1.0f, 7.0f); // U = 7I - AZ
    launch_gemm(AZ, U, T,  256, 256, 256, -1.0f, 11.0f, stream);      // T = 11I - AZ@U
    launch_gemm(AZ, T, U,  256, 256, 256, -1.0f, 13.0f, stream);      // U = 13I - AZ@T
    launch_gemm(zc, U, zn, 256, 256, 256, 0.25f,  0.0f, stream);      // z' = 0.25 z@U
    float* tmp = zc; zc = zn; zn = tmp;
  }

  k_a3v <<<128, 128, 0, stream>>>(qlh, kh, vh, a3v);
  launch_gemm(zc, a3v, w2, 256, 64, 256, 1.0f, 0.0f, stream);         // w2 = pinv @ (a3@v)
  k_cvt <<<(BH_*M_*DH_)/256, 256, 0, stream>>>(w2, w2h, BH_*M_*DH_);
  k_a1w <<<4096, 128, 0, stream>>>(qh, klh, w2h, out1);
  k_conv<<<(ROWS_*D_)/256, 256, 0, stream>>>(out1, vh, conv_w, att);
  k_final<<<2048, 256, 0, stream>>>(att, woh, b_out, x, outp);
}
